// RelativeMultiHeadAttention_25503515803851
// MI455X (gfx1250) — matrix operand order repaired and checked by reading, not compiled, not run on hardware
//
#include <hip/hip_runtime.h>
#include <hip/hip_bf16.h>

// ---------------- problem constants ----------------
#define BB   4
#define NN   2048
#define DIMM 1024
#define HH   8
#define DKK  64
#define HD   512          // HH*DKK
#define SCALEQ 0.125f     // 64^-0.5

typedef __attribute__((ext_vector_type(16))) _Float16 v16h;
typedef __attribute__((ext_vector_type(8)))  float    v8f;

struct __align__(16) Half8 { _Float16 h[8]; };

// ---------------- WMMA helpers ----------------
__device__ __forceinline__ v8f wmma_f16(v16h a, v16h b, v8f c) {
  // D = A(16x32 f16) * B(32x16 f16) + C(16x16 f32)
  return __builtin_amdgcn_wmma_f32_16x16x32_f16(
      /*neg_a=*/false, a, /*neg_b=*/false, b,
      /*c_mod=*/(short)0, c, /*reuse_a=*/false, /*reuse_b=*/false);
}

// A fragment: lane<16 holds row M=lane, K = {khi..khi+7, 16+khi..16+khi+7}, khi=(lane>>4)*8
__device__ __forceinline__ v16h load_a_f16(const _Float16* base, int ld, int row0, int k, int lane) {
  const int r   = row0 + (lane & 15);
  const int khi = (lane >> 4) * 8;
  const _Float16* p = base + (size_t)r * ld + k + khi;
  Half8 lo = *reinterpret_cast<const Half8*>(p);
  Half8 hi = *reinterpret_cast<const Half8*>(p + 16);
  v16h a;
#pragma unroll
  for (int i = 0; i < 8; ++i) { a[i] = lo.h[i]; a[8 + i] = hi.h[i]; }
  return a;
}

// B fragment: lane holds column N=lane&15, K = {kb..kb+7, 16+kb..16+kb+7}, kb=(lane>>4)*8.
// base is stored column-major for the GEMM: base[col*ld + k].
__device__ __forceinline__ v16h load_b_f16(const _Float16* base, int ld, int col0, int k, int lane) {
  const int c  = col0 + (lane & 15);
  const int kb = (lane >> 4) * 8;
  const _Float16* p = base + (size_t)c * ld + k + kb;
  Half8 lo = *reinterpret_cast<const Half8*>(p);
  Half8 hi = *reinterpret_cast<const Half8*>(p + 16);
  v16h b;
#pragma unroll
  for (int i = 0; i < 8; ++i) { b[i] = lo.h[i]; b[8 + i] = hi.h[i]; }
  return b;
}

// A fragment from f32 matrix, converted to f16 on the fly (attn values in [0,1])
__device__ __forceinline__ v16h load_a_f32(const float* base, int ld, int row0, int k, int lane) {
  const int r   = row0 + (lane & 15);
  const int khi = (lane >> 4) * 8;
  const float* p = base + (size_t)r * ld + k + khi;
  float4 a0 = *reinterpret_cast<const float4*>(p);
  float4 a1 = *reinterpret_cast<const float4*>(p + 4);
  float4 b0 = *reinterpret_cast<const float4*>(p + 16);
  float4 b1 = *reinterpret_cast<const float4*>(p + 20);
  float f[16] = {a0.x, a0.y, a0.z, a0.w, a1.x, a1.y, a1.z, a1.w,
                 b0.x, b0.y, b0.z, b0.w, b1.x, b1.y, b1.z, b1.w};
  v16h a;
#pragma unroll
  for (int i = 0; i < 16; ++i) a[i] = (_Float16)f[i];
  return a;
}

// ---------------- conversion kernels ----------------
__global__ void __launch_bounds__(256) k_cvt_x(const float* __restrict__ x, _Float16* __restrict__ xh) {
  const size_t i = (size_t)blockIdx.x * 256 + threadIdx.x;   // 8388608 elems
  xh[i] = (_Float16)x[i];
}

// y = 0..2: Wq/Wk/Wv [1024,512] -> WT [512][1024];  y = 3: Wo [512,1024] -> WoT [1024][512]
__global__ void __launch_bounds__(256) k_cvt_w(const float* __restrict__ Wq, const float* __restrict__ Wk,
                                               const float* __restrict__ Wv, const float* __restrict__ Wo,
                                               _Float16* __restrict__ wqT, _Float16* __restrict__ wkT,
                                               _Float16* __restrict__ wvT, _Float16* __restrict__ woT) {
  const int i = blockIdx.x * 256 + threadIdx.x;              // 524288 per matrix
  const int y = blockIdx.y;
  if (y < 3) {
    const float* src = (y == 0) ? Wq : (y == 1) ? Wk : Wv;
    _Float16*    dst = (y == 0) ? wqT : (y == 1) ? wkT : wvT;
    const int c = i >> 10, r = i & 1023;                     // dst[c][r] = src[r][c]
    dst[i] = (_Float16)src[r * HD + c];
  } else {
    const int c = i >> 9, r = i & 511;                       // woT[c][r] = Wo[r][c]
    woT[i] = (_Float16)Wo[r * DIMM + c];
  }
}

// ---------------- QKV projection: [8192,1024] x [1024,512], ping-pong pipelined ----------------
__global__ void __launch_bounds__(128) k_qkv(const _Float16* __restrict__ xh,
                                             const _Float16* __restrict__ wqT,
                                             const _Float16* __restrict__ wkT,
                                             const _Float16* __restrict__ wvT,
                                             _Float16* __restrict__ qh,
                                             _Float16* __restrict__ kh,
                                             _Float16* __restrict__ vt) {
  const int lane = threadIdx.x & 31, wave = threadIdx.x >> 5;
  const int z = blockIdx.z;
  const _Float16* wt = (z == 0) ? wqT : (z == 1) ? wkT : wvT;
  const int m0 = blockIdx.y * 64 + wave * 16;
  const int n0 = blockIdx.x * 64;
  const v8f zero = {0.f, 0.f, 0.f, 0.f, 0.f, 0.f, 0.f, 0.f};
  v8f acc[4] = {zero, zero, zero, zero};

  v16h a0 = load_a_f16(xh, DIMM, m0, 0, lane);
  v16h b0[4], b1[4];
  v16h a1;
#pragma unroll
  for (int nt = 0; nt < 4; ++nt) b0[nt] = load_b_f16(wt, DIMM, n0 + nt * 16, 0, lane);

  int k = 32;
  for (; k < DIMM - 32; k += 64) {
    a1 = load_a_f16(xh, DIMM, m0, k, lane);
#pragma unroll
    for (int nt = 0; nt < 4; ++nt) b1[nt] = load_b_f16(wt, DIMM, n0 + nt * 16, k, lane);
    __builtin_prefetch(xh + (size_t)(m0 + (lane & 15)) * DIMM + k + 128, 0, 1);
#pragma unroll
    for (int nt = 0; nt < 4; ++nt) acc[nt] = wmma_f16(a0, b0[nt], acc[nt]);

    a0 = load_a_f16(xh, DIMM, m0, k + 32, lane);
#pragma unroll
    for (int nt = 0; nt < 4; ++nt) b0[nt] = load_b_f16(wt, DIMM, n0 + nt * 16, k + 32, lane);
#pragma unroll
    for (int nt = 0; nt < 4; ++nt) acc[nt] = wmma_f16(a1, b1[nt], acc[nt]);
  }
  // leftover step at k == DIMM-32
  a1 = load_a_f16(xh, DIMM, m0, DIMM - 32, lane);
#pragma unroll
  for (int nt = 0; nt < 4; ++nt) b1[nt] = load_b_f16(wt, DIMM, n0 + nt * 16, DIMM - 32, lane);
#pragma unroll
  for (int nt = 0; nt < 4; ++nt) acc[nt] = wmma_f16(a0, b0[nt], acc[nt]);
#pragma unroll
  for (int nt = 0; nt < 4; ++nt) acc[nt] = wmma_f16(a1, b1[nt], acc[nt]);

  const int colL = lane & 15, rowHi = (lane >> 4) * 8;
#pragma unroll
  for (int nt = 0; nt < 4; ++nt) {
#pragma unroll
    for (int r = 0; r < 8; ++r) {
      const float v = acc[nt][r];
      const int mg = m0 + rowHi + r;                 // global token row 0..8191
      const int ng = n0 + nt * 16 + colL;            // 0..511
      const int bI = mg >> 11, tok = mg & (NN - 1);
      const int hI = ng >> 6,  d   = ng & 63;
      const int bh = bI * HH + hI;
      if (z == 0)      qh[((size_t)(bh * NN + tok)) * DKK + d] = (_Float16)(v * SCALEQ);
      else if (z == 1) kh[((size_t)(bh * NN + tok)) * DKK + d] = (_Float16)v;
      else             vt[((size_t)(bh * DKK + d)) * NN + tok] = (_Float16)v;  // V transposed
    }
  }
}

// ---------------- logits: S = Qs * K^T per (b,h); all loads hoisted, 8 back-to-back WMMAs ----------------
__global__ void __launch_bounds__(128) k_logits(const _Float16* __restrict__ qh,
                                                const _Float16* __restrict__ kh,
                                                float* __restrict__ attn) {
  const int lane = threadIdx.x & 31, wave = threadIdx.x >> 5;
  const int bh = blockIdx.z;
  const _Float16* qb = qh + (size_t)bh * NN * DKK;
  const _Float16* kb = kh + (size_t)bh * NN * DKK;   // K rows == B columns (contiguous in k)
  const int m0 = blockIdx.y * 64 + wave * 16;
  const int n0 = blockIdx.x * 64;
  const v8f zero = {0.f, 0.f, 0.f, 0.f, 0.f, 0.f, 0.f, 0.f};
  v8f acc[4] = {zero, zero, zero, zero};

  v16h a0 = load_a_f16(qb, DKK, m0, 0,  lane);
  v16h a1 = load_a_f16(qb, DKK, m0, 32, lane);
  v16h b0[4], b1[4];
#pragma unroll
  for (int nt = 0; nt < 4; ++nt) {
    b0[nt] = load_b_f16(kb, DKK, n0 + nt * 16, 0,  lane);
    b1[nt] = load_b_f16(kb, DKK, n0 + nt * 16, 32, lane);
  }
#pragma unroll
  for (int nt = 0; nt < 4; ++nt) acc[nt] = wmma_f16(a0, b0[nt], acc[nt]);
#pragma unroll
  for (int nt = 0; nt < 4; ++nt) acc[nt] = wmma_f16(a1, b1[nt], acc[nt]);

  float* out = attn + (size_t)bh * NN * NN;
  const int colL = lane & 15, rowHi = (lane >> 4) * 8;
#pragma unroll
  for (int nt = 0; nt < 4; ++nt)
#pragma unroll
    for (int r = 0; r < 8; ++r)
      out[(size_t)(m0 + rowHi + r) * NN + (n0 + nt * 16 + colL)] = acc[nt][r];
}

// ---------------- row softmax in place over attn ----------------
__global__ void __launch_bounds__(256) k_softmax(float* __restrict__ attn) {
  __shared__ float red[256];
  const int t = threadIdx.x;
  float* row = attn + (size_t)blockIdx.x * NN;
  float v[8];
  float m = -3.4e38f;
#pragma unroll
  for (int i = 0; i < 8; ++i) { v[i] = row[t + i * 256]; m = fmaxf(m, v[i]); }
  red[t] = m; __syncthreads();
  for (int s = 128; s > 0; s >>= 1) { if (t < s) red[t] = fmaxf(red[t], red[t + s]); __syncthreads(); }
  m = red[0]; __syncthreads();
  float sum = 0.f;
#pragma unroll
  for (int i = 0; i < 8; ++i) { v[i] = __expf(v[i] - m); sum += v[i]; }
  red[t] = sum; __syncthreads();
  for (int s = 128; s > 0; s >>= 1) { if (t < s) red[t] += red[t + s]; __syncthreads(); }
  const float inv = 1.0f / red[0];
#pragma unroll
  for (int i = 0; i < 8; ++i) row[t + i * 256] = v[i] * inv;
}

// ---------------- O = attn @ V per (b,h), K = 2048, ping-pong pipelined ----------------
__global__ void __launch_bounds__(128) k_pv(const float* __restrict__ attn,
                                            const _Float16* __restrict__ vt,
                                            _Float16* __restrict__ oh) {
  const int lane = threadIdx.x & 31, wave = threadIdx.x >> 5;
  const int bh = blockIdx.y;
  const float*    ab = attn + (size_t)bh * NN * NN;
  const _Float16* vb = vt   + (size_t)bh * DKK * NN;  // [d][n], contiguous in k
  const int m0 = blockIdx.x * 64 + wave * 16;
  const v8f zero = {0.f, 0.f, 0.f, 0.f, 0.f, 0.f, 0.f, 0.f};
  v8f acc[4] = {zero, zero, zero, zero};

  v16h a0 = load_a_f32(ab, NN, m0, 0, lane);
  v16h b0[4], b1[4];
  v16h a1;
#pragma unroll
  for (int nt = 0; nt < 4; ++nt) b0[nt] = load_b_f16(vb, NN, nt * 16, 0, lane);

  int k = 32;
  for (; k < NN - 32; k += 64) {
    a1 = load_a_f32(ab, NN, m0, k, lane);
#pragma unroll
    for (int nt = 0; nt < 4; ++nt) b1[nt] = load_b_f16(vb, NN, nt * 16, k, lane);
    __builtin_prefetch(ab + (size_t)(m0 + (lane & 15)) * NN + k + 128, 0, 1);
#pragma unroll
    for (int nt = 0; nt < 4; ++nt) acc[nt] = wmma_f16(a0, b0[nt], acc[nt]);

    a0 = load_a_f32(ab, NN, m0, k + 32, lane);
#pragma unroll
    for (int nt = 0; nt < 4; ++nt) b0[nt] = load_b_f16(vb, NN, nt * 16, k + 32, lane);
#pragma unroll
    for (int nt = 0; nt < 4; ++nt) acc[nt] = wmma_f16(a1, b1[nt], acc[nt]);
  }
  // leftover step at k == NN-32
  a1 = load_a_f32(ab, NN, m0, NN - 32, lane);
#pragma unroll
  for (int nt = 0; nt < 4; ++nt) b1[nt] = load_b_f16(vb, NN, nt * 16, NN - 32, lane);
#pragma unroll
  for (int nt = 0; nt < 4; ++nt) acc[nt] = wmma_f16(a0, b0[nt], acc[nt]);
#pragma unroll
  for (int nt = 0; nt < 4; ++nt) acc[nt] = wmma_f16(a1, b1[nt], acc[nt]);

  const int bI = bh >> 3, hI = bh & 7;
  const int colL = lane & 15, rowHi = (lane >> 4) * 8;
#pragma unroll
  for (int nt = 0; nt < 4; ++nt)
#pragma unroll
    for (int r = 0; r < 8; ++r) {
      const int tok = m0 + rowHi + r;
      const int d   = nt * 16 + colL;
      oh[((size_t)(bI * NN + tok)) * HD + hI * DKK + d] = (_Float16)acc[nt][r];
    }
}

// ---------------- out = O @ Wo + bo : [8192,512] x [512,1024], ping-pong pipelined ----------------
__global__ void __launch_bounds__(128) k_out(const _Float16* __restrict__ oh,
                                             const _Float16* __restrict__ woT,
                                             const float* __restrict__ bo,
                                             float* __restrict__ outp) {
  const int lane = threadIdx.x & 31, wave = threadIdx.x >> 5;
  const int m0 = blockIdx.y * 64 + wave * 16;
  const int n0 = blockIdx.x * 64;
  const v8f zero = {0.f, 0.f, 0.f, 0.f, 0.f, 0.f, 0.f, 0.f};
  v8f acc[4] = {zero, zero, zero, zero};

  v16h a0 = load_a_f16(oh, HD, m0, 0, lane);
  v16h b0[4], b1[4];
  v16h a1;
#pragma unroll
  for (int nt = 0; nt < 4; ++nt) b0[nt] = load_b_f16(woT, HD, n0 + nt * 16, 0, lane);

  int k = 32;
  for (; k < HD - 32; k += 64) {
    a1 = load_a_f16(oh, HD, m0, k, lane);
#pragma unroll
    for (int nt = 0; nt < 4; ++nt) b1[nt] = load_b_f16(woT, HD, n0 + nt * 16, k, lane);
#pragma unroll
    for (int nt = 0; nt < 4; ++nt) acc[nt] = wmma_f16(a0, b0[nt], acc[nt]);

    a0 = load_a_f16(oh, HD, m0, k + 32, lane);
#pragma unroll
    for (int nt = 0; nt < 4; ++nt) b0[nt] = load_b_f16(woT, HD, n0 + nt * 16, k + 32, lane);
#pragma unroll
    for (int nt = 0; nt < 4; ++nt) acc[nt] = wmma_f16(a1, b1[nt], acc[nt]);
  }
  // leftover step at k == HD-32
  a1 = load_a_f16(oh, HD, m0, HD - 32, lane);
#pragma unroll
  for (int nt = 0; nt < 4; ++nt) b1[nt] = load_b_f16(woT, HD, n0 + nt * 16, HD - 32, lane);
#pragma unroll
  for (int nt = 0; nt < 4; ++nt) acc[nt] = wmma_f16(a0, b0[nt], acc[nt]);
#pragma unroll
  for (int nt = 0; nt < 4; ++nt) acc[nt] = wmma_f16(a1, b1[nt], acc[nt]);

  const int colL = lane & 15, rowHi = (lane >> 4) * 8;
#pragma unroll
  for (int nt = 0; nt < 4; ++nt)
#pragma unroll
    for (int r = 0; r < 8; ++r) {
      const int mg = m0 + rowHi + r;
      const int ng = n0 + nt * 16 + colL;
      outp[(size_t)mg * DIMM + ng] = acc[nt][r] + bo[ng];
    }
}

// ---------------- launcher ----------------
extern "C" void kernel_launch(void* const* d_in, const int* in_sizes, int n_in,
                              void* d_out, int out_size, void* d_ws, size_t ws_size,
                              hipStream_t stream) {
  (void)in_sizes; (void)n_in; (void)out_size; (void)ws_size;
  const float* x  = (const float*)d_in[0];
  const float* Wq = (const float*)d_in[1];
  const float* Wk = (const float*)d_in[2];
  const float* Wv = (const float*)d_in[3];
  const float* Wo = (const float*)d_in[4];
  const float* bo = (const float*)d_in[5];

  // workspace layout (halves)
  _Float16* wsh = (_Float16*)d_ws;
  _Float16* xh  = wsh;                          // 8192*1024
  _Float16* wqT = xh  + (size_t)8192 * 1024;    // 512*1024
  _Float16* wkT = wqT + (size_t)512 * 1024;
  _Float16* wvT = wkT + (size_t)512 * 1024;
  _Float16* woT = wvT + (size_t)512 * 1024;     // 1024*512
  _Float16* qh  = woT + (size_t)1024 * 512;     // 32*2048*64 each
  _Float16* kh  = qh  + (size_t)32 * 2048 * 64;
  _Float16* vt  = kh  + (size_t)32 * 2048 * 64;
  _Float16* oh  = vt  + (size_t)32 * 2048 * 64; // 8192*512

  float* outp  = (float*)d_out;
  float* attnp = outp + (size_t)BB * NN * DIMM; // attn region of d_out

  k_cvt_x  <<<32768, 256, 0, stream>>>(x, xh);
  k_cvt_w  <<<dim3(2048, 4), 256, 0, stream>>>(Wq, Wk, Wv, Wo, wqT, wkT, wvT, woT);
  k_qkv    <<<dim3(8, 128, 3), 128, 0, stream>>>(xh, wqT, wkT, wvT, qh, kh, vt);
  k_logits <<<dim3(32, 32, 32), 128, 0, stream>>>(qh, kh, attnp);
  k_softmax<<<BB * HH * NN, 256, 0, stream>>>(attnp);
  k_pv     <<<dim3(32, 32), 128, 0, stream>>>(attnp, vt, oh);
  k_out    <<<dim3(16, 128), 128, 0, stream>>>(oh, woT, bo, outp);
}
